// MultiHeadLatentAttention_48352741819171
// MI455X (gfx1250) — compile-verified, matrix operand order checked by reading, not run on hardware
//
#include <hip/hip_runtime.h>
#include <stdint.h>

// ---------------------------------------------------------------------------
// MLA forward for MI455X (gfx1250, wave32, WMMA bf16 16x16x32 f32-accum)
// ---------------------------------------------------------------------------

#define B_  2
#define S_  2048
#define D_  2048
#define H_  16
#define QR_ 768
#define KVR_ 512
#define DN_ 128
#define DR_ 64
#define DV_ 128
#define BS_ (B_ * S_)
#define EPS_ 1e-6f

typedef __attribute__((ext_vector_type(16))) __bf16 v16bf;
typedef __attribute__((ext_vector_type(8)))  float  v8f;

// Types for the async global->LDS builtin: param 1 prints as
// 'int __vector_size__(4*sizeof(int)) __device__ *'  (= int4 addrspace(1)*)
typedef int v4i_ __attribute__((__vector_size__(16)));
typedef __attribute__((address_space(1))) v4i_ g_v4i;   // global
typedef __attribute__((address_space(3))) v4i_ l_v4i;   // LDS

union BFrag {
    v16bf v;
    uint4 q[2];
};

__device__ __forceinline__ uint16_t f2bf(float f) {
    union { float f; uint32_t u; } c; c.f = f;
    uint32_t u = c.u;
    uint32_t r = u + 0x7fffu + ((u >> 16) & 1u);   // round-to-nearest-even
    return (uint16_t)(r >> 16);
}
__device__ __forceinline__ float bf2f(uint16_t b) {
    union { uint32_t u; float f; } c; c.u = ((uint32_t)b) << 16;
    return c.f;
}

// ---------------------------------------------------------------------------
// Async global->LDS staging (GLOBAL_LOAD_ASYNC_TO_LDS_B128, ASYNCcnt-tracked).
// Guarded by __has_builtin; falls back to a plain 16B copy.
// ---------------------------------------------------------------------------

#if __has_builtin(__builtin_amdgcn_global_load_async_to_lds_b128)
#define HAVE_ASYNC_LDS 1
#else
#define HAVE_ASYNC_LDS 0
#endif

__device__ __forceinline__ void stage16(uint16_t* l, const uint16_t* g) {
#if HAVE_ASYNC_LDS
    // global generic addr == addrspace(1) addr; generic LDS pointer low 32
    // bits == LDS byte offset (ISA 10.2 aperture map)
    __builtin_amdgcn_global_load_async_to_lds_b128(
        (g_v4i*)(uintptr_t)g,
        (l_v4i*)(uint32_t)(uintptr_t)l,
        /*offset=*/0, /*cpol=*/0);
#else
    *(uint4*)l = *(const uint4*)g;
#endif
}

__device__ __forceinline__ void stage_wait() {
#if HAVE_ASYNC_LDS
#if __has_builtin(__builtin_amdgcn_s_wait_asynccnt)
    __builtin_amdgcn_s_wait_asynccnt(0);
#else
    asm volatile("s_wait_asynccnt 0" ::: "memory");
#endif
#endif
}

// Load one 16x32 bf16 A/B-fragment row (per-lane view) from a row-major
// (row, 32 K) region.  lanehi = (lane >= 16).  Matches ISA 7.12.2:
//   lanes 0-15 : K = {0..7, 16..23};  lanes 16-31 : K = {8..15, 24..31}
__device__ __forceinline__ BFrag load_frag_row(const uint16_t* rowptr, int lanehi) {
    BFrag f;
    const int kb = lanehi * 8;
    f.q[0] = *(const uint4*)(rowptr + kb);
    f.q[1] = *(const uint4*)(rowptr + kb + 16);
    return f;
}

__device__ __forceinline__ v8f wmma_bf16(const BFrag& a, const BFrag& b, v8f c) {
    return __builtin_amdgcn_wmma_f32_16x16x32_bf16(
        /*neg_a=*/false, a.v, /*neg_b=*/false, b.v,
        /*c_mod=*/(short)0, c, /*reuse_a=*/false, /*reuse_b=*/false);
}

// ---------------------------------------------------------------------------
// Elementwise helpers
// ---------------------------------------------------------------------------

__global__ __launch_bounds__(256) void cvt_f32_bf16_kernel(
    const float* __restrict__ in, uint16_t* __restrict__ out, int n) {
    int i = blockIdx.x * 256 + threadIdx.x;
    if (i < n) out[i] = f2bf(in[i]);
}

// out[c*R + r] = bf16(in[r*C + c])  : convert + transpose (weights -> Bt[N][K])
__global__ __launch_bounds__(256) void cvt_transpose_kernel(
    const float* __restrict__ in, uint16_t* __restrict__ out, int R, int C) {
    int i = blockIdx.x * 256 + threadIdx.x;
    if (i >= R * C) return;
    int r = i / C, c = i % C;
    out[(size_t)c * R + r] = f2bf(in[(size_t)r * C + c]);
}

// ---------------------------------------------------------------------------
// Generic GEMM: C[M,N] = A[M,K] (bf16, row-major) x Bt[N,K] (bf16, N-major)
// Block tile 128x128, 8 waves (2x4), wave tile 64x32, K-step 32.
// Staging via async global->LDS copies.
// ---------------------------------------------------------------------------

#define LDS_PAD 48   // elements per LDS row (32 used + pad); 96B, 16B aligned

template <bool OUT_BF16>
__global__ __launch_bounds__(256) void gemm_bf16_kernel(
    const uint16_t* __restrict__ A, const uint16_t* __restrict__ Bt,
    void* __restrict__ Cout, int M, int N, int K) {
    __shared__ uint16_t lA[128 * LDS_PAD];
    __shared__ uint16_t lB[128 * LDS_PAD];

    const int tid    = threadIdx.x;
    const int lane   = tid & 31;
    const int wid    = tid >> 5;
    const int lanehi = lane >> 4;
    const int lanelo = lane & 15;
    const int m0     = blockIdx.y * 128;
    const int n0     = blockIdx.x * 128;
    const int waveM  = wid >> 2;   // 0..1 (64 rows each)
    const int waveN  = wid & 3;    // 0..3 (32 cols each)

    v8f acc[4][2];
#pragma unroll
    for (int mi = 0; mi < 4; ++mi)
#pragma unroll
        for (int ni = 0; ni < 2; ++ni) acc[mi][ni] = {};

    for (int k0 = 0; k0 < K; k0 += 32) {
        // ---- stage 128x32 A tile and 128x32 Bt tile into LDS (async) ----
#pragma unroll
        for (int i = 0; i < 2; ++i) {
            int chunk = tid * 2 + i;       // 0..511
            int row   = chunk >> 2;        // 0..127
            int c     = (chunk & 3) * 8;   // bf16 element offset, 16B chunks
            stage16(lA + row * LDS_PAD + c, A + (size_t)(m0 + row) * K + k0 + c);
            // OOB Bt rows are skipped: their (stale) LDS contents only reach
            // accumulator columns that are never stored.
            if (n0 + row < N)
                stage16(lB + row * LDS_PAD + c, Bt + (size_t)(n0 + row) * K + k0 + c);
        }
        stage_wait();
        __syncthreads();

        BFrag af[4], bfrag[2];
#pragma unroll
        for (int mi = 0; mi < 4; ++mi)
            af[mi] = load_frag_row(lA + (waveM * 64 + mi * 16 + lanelo) * LDS_PAD, lanehi);
#pragma unroll
        for (int ni = 0; ni < 2; ++ni)
            bfrag[ni] = load_frag_row(lB + (waveN * 32 + ni * 16 + lanelo) * LDS_PAD, lanehi);

#pragma unroll
        for (int mi = 0; mi < 4; ++mi)
#pragma unroll
            for (int ni = 0; ni < 2; ++ni)
                acc[mi][ni] = wmma_bf16(af[mi], bfrag[ni], acc[mi][ni]);
        __syncthreads();
    }

    // ---- store: C/D layout, VGPR r <-> row r + 8*lanehi, col = lanelo ----
#pragma unroll
    for (int mi = 0; mi < 4; ++mi)
#pragma unroll
        for (int ni = 0; ni < 2; ++ni)
#pragma unroll
            for (int r = 0; r < 8; ++r) {
                int m = m0 + waveM * 64 + mi * 16 + r + 8 * lanehi;
                int n = n0 + waveN * 32 + ni * 16 + lanelo;
                if (n < N) {
                    float v = acc[mi][ni][r];
                    if (OUT_BF16)
                        ((uint16_t*)Cout)[(size_t)m * N + n] = f2bf(v);
                    else
                        ((float*)Cout)[(size_t)m * N + n] = v;
                }
            }
}

// ---------------------------------------------------------------------------
// RMS norm over rows of length R (bf16 in/out, f32 accumulation)
// ---------------------------------------------------------------------------

__global__ __launch_bounds__(256) void rmsnorm_kernel(
    const uint16_t* __restrict__ in, const float* __restrict__ w,
    uint16_t* __restrict__ out, int R) {
    __shared__ float red[256];
    const int row = blockIdx.x;
    const uint16_t* x = in + (size_t)row * R;
    float s = 0.f;
    for (int i = threadIdx.x; i < R; i += 256) {
        float v = bf2f(x[i]);
        s += v * v;
    }
    red[threadIdx.x] = s;
    __syncthreads();
    for (int o = 128; o > 0; o >>= 1) {
        if (threadIdx.x < (unsigned)o) red[threadIdx.x] += red[threadIdx.x + o];
        __syncthreads();
    }
    float inv = rsqrtf(red[0] / (float)R + EPS_);
    for (int i = threadIdx.x; i < R; i += 256)
        out[(size_t)row * R + i] = f2bf(bf2f(x[i]) * inv * w[i]);
}

// ---------------------------------------------------------------------------
// RoPE on interleaved pairs: layout [BS, Hh*64], position = row % S
// ---------------------------------------------------------------------------

__global__ __launch_bounds__(256) void rope_kernel(
    const uint16_t* __restrict__ in, uint16_t* __restrict__ out,
    int Hh, int total) {
    int idx = blockIdx.x * 256 + threadIdx.x;
    if (idx >= total) return;            // total = BS*Hh*32
    int i   = idx & 31;                  // pair index 0..31
    int rem = idx >> 5;                  // bs*Hh + h
    int h   = rem % Hh;
    int bs  = rem / Hh;
    int pos = bs % S_;
    // freq = pos * 10000^(-(2i)/64)
    float f  = (float)pos * __expf(-((float)(2 * i) / 64.0f) * 9.210340372f);
    float cs = __cosf(f), sn = __sinf(f);
    size_t base = (size_t)bs * (Hh * 64) + h * 64 + 2 * i;
    float x1 = bf2f(in[base]), x2 = bf2f(in[base + 1]);
    out[base]     = f2bf(x1 * cs - x2 * sn);
    out[base + 1] = f2bf(x1 * sn + x2 * cs);
}

// ---------------------------------------------------------------------------
// Build Vt cache: vt[b,h,dv,s] = kvu[bs, h*256 + 128 + dv]
// ---------------------------------------------------------------------------

__global__ __launch_bounds__(256) void vtrans_kernel(
    const uint16_t* __restrict__ kvu, uint16_t* __restrict__ vt) {
    int idx = blockIdx.x * 256 + threadIdx.x;  // over B*H*DV*S
    if (idx >= B_ * H_ * DV_ * S_) return;
    int s  = idx % S_;
    int t  = idx / S_;
    int dv = t % DV_;
    t /= DV_;
    int h = t % H_;
    int b = t / H_;
    size_t src = ((size_t)(b * S_ + s) * H_ + h) * 256 + 128 + dv;
    vt[idx] = kvu[src];
}

// ---------------------------------------------------------------------------
// Block-cooperative flash attention.
// Block = 8 waves = 8 consecutive 16-query tiles (128 queries) of one (b,h).
// Per 32-key chunk, the block stages K (32x192) and Vt (128x32) into LDS once
// (async global->LDS); every wave then reads WMMA fragments from LDS.
// 12 score WMMAs + online softmax + 8 PV WMMAs per active wave per chunk.
// ---------------------------------------------------------------------------

#define KPAD 208   // K-tile row stride in elements (192 + 16); 416B, 16B-aligned

__global__ __launch_bounds__(256) void mla_attn_kernel(
    const uint16_t* __restrict__ qn,   // [BS, H*128]
    const uint16_t* __restrict__ qpr,  // [BS, H*64]  (rope applied)
    const uint16_t* __restrict__ kvu,  // [BS, H*256] (first 128/head = k_nope)
    const uint16_t* __restrict__ krr,  // [BS, 64]    (rope applied, shared over h)
    const uint16_t* __restrict__ vt,   // [B, H, 128, S]
    uint16_t* __restrict__ out) {      // [BS, H*128]
    __shared__ uint16_t kl[32 * KPAD];           // 13312 B: 32 keys x 192 head-dim
    __shared__ uint16_t vl[128 * LDS_PAD];       // 12288 B: 128 v-dims x 32 keys
    __shared__ uint16_t plds[8][16 * LDS_PAD];   // 12288 B: per-wave P transpose

    const int tid    = threadIdx.x;
    const int lane   = tid & 31;
    const int wid    = tid >> 5;
    const int lanehi = lane >> 4;
    const int lanelo = lane & 15;

    const int grp = blockIdx.x & 15;   // 16 query groups of 128 per (b,h)
    const int bh  = blockIdx.x >> 4;
    const int h   = bh % H_;
    const int b   = bh / H_;
    const int q0  = grp * 128 + wid * 16;   // this wave's query tile
    const int kend_blk = grp * 128 + 128;   // keys needed by the whole block

    // ---- preload the 6 Q fragments (192 = 6 x 32 K) ----
    BFrag qf[6];
    {
        size_t bs = (size_t)b * S_ + q0 + lanelo;
        const uint16_t* qnrow = qn + bs * (H_ * 128) + h * 128;
        const uint16_t* qprow = qpr + bs * (H_ * 64) + h * 64;
#pragma unroll
        for (int c = 0; c < 4; ++c) qf[c] = load_frag_row(qnrow + c * 32, lanehi);
#pragma unroll
        for (int c = 0; c < 2; ++c) qf[4 + c] = load_frag_row(qprow + c * 32, lanehi);
    }

    v8f accO[8];
#pragma unroll
    for (int nt = 0; nt < 8; ++nt) accO[nt] = {};
    float rmax[8], rsum[8];
#pragma unroll
    for (int r = 0; r < 8; ++r) { rmax[r] = -1e30f; rsum[r] = 0.f; }

    const float scale = 0.0721687836f;  // 1/sqrt(192)
    const uint16_t* vbase = vt + ((size_t)(b * H_ + h) * DV_) * S_;

    for (int c = 0; c < kend_blk; c += 32) {
        // ---- cooperative stage: K chunk = 32 keys x 192 = 768 x 16B chunks --
#pragma unroll
        for (int i = 0; i < 3; ++i) {
            int ci   = tid + i * 256;       // 0..767
            int key  = ci / 24;             // 24 x 16B chunks per key row
            int part = ci % 24;
            size_t bs = (size_t)b * S_ + c + key;
            const uint16_t* src =
                (part < 16) ? kvu + bs * (size_t)(H_ * 256) + h * 256 + part * 8
                            : krr + bs * 64 + (part - 16) * 8;
            stage16(kl + key * KPAD + part * 8, src);
        }
        // ---- cooperative stage: Vt chunk = 128 rows x 32 keys = 512 chunks --
#pragma unroll
        for (int i = 0; i < 2; ++i) {
            int ci  = tid + i * 256;        // 0..511
            int row = ci >> 2;              // v-dim 0..127
            int p   = ci & 3;
            stage16(vl + row * LDS_PAD + p * 8,
                    vbase + (size_t)row * S_ + c + p * 8);
        }
        stage_wait();
        __syncthreads();

        if (c < q0 + 16) {  // causal: this wave still has keys in this chunk
            // ---- scores for 2 key subtiles of 16, K fragments from LDS ----
            v8f s0 = {}, s1 = {};
#pragma unroll
            for (int st = 0; st < 2; ++st) {
                const uint16_t* krow = kl + (st * 16 + lanelo) * KPAD;
                v8f sacc = st ? s1 : s0;
#pragma unroll
                for (int cc = 0; cc < 6; ++cc) {
                    BFrag kf = load_frag_row(krow + cc * 32, lanehi);
                    sacc = wmma_bf16(qf[cc], kf, sacc);
                }
                if (st) s1 = sacc; else s0 = sacc;
            }

            // ---- online softmax (rows r + 8*lanehi, cols = lanelo) ----
            const bool needMask = (c + 32 > q0);
            uint16_t* pl = plds[wid];
#pragma unroll
            for (int r = 0; r < 8; ++r) {
                int m = r + 8 * lanehi;
                int q = q0 + m;
                float v0 = s0[r] * scale, v1 = s1[r] * scale;
                if (needMask) {
                    if (c + lanelo > q)      v0 = -1e30f;
                    if (c + 16 + lanelo > q) v1 = -1e30f;
                }
                float mx = fmaxf(v0, v1);
#pragma unroll
                for (int o = 1; o < 16; o <<= 1) mx = fmaxf(mx, __shfl_xor(mx, o, 32));
                float nm   = fmaxf(rmax[r], mx);
                float corr = __expf(rmax[r] - nm);
                float p0   = __expf(v0 - nm);
                float p1   = __expf(v1 - nm);
                float ps   = p0 + p1;
#pragma unroll
                for (int o = 1; o < 16; o <<= 1) ps += __shfl_xor(ps, o, 32);
                rsum[r] = rsum[r] * corr + ps;
                rmax[r] = nm;
#pragma unroll
                for (int nt = 0; nt < 8; ++nt) accO[nt][r] *= corr;
                // transpose P through per-wave LDS into A-fragment layout
                pl[m * LDS_PAD + lanelo]      = f2bf(p0);
                pl[m * LDS_PAD + 16 + lanelo] = f2bf(p1);
            }
            asm volatile("s_wait_dscnt 0" ::: "memory");

            BFrag pf = load_frag_row(pl + lanelo * LDS_PAD, lanehi);

            // ---- PV: 8 N-tiles of 16 v-dims, K = 32 keys, V from LDS ----
#pragma unroll
            for (int nt = 0; nt < 8; ++nt) {
                BFrag vf = load_frag_row(vl + (nt * 16 + lanelo) * LDS_PAD, lanehi);
                accO[nt] = wmma_bf16(pf, vf, accO[nt]);
            }
        }
        __syncthreads();   // protect kl/vl before next chunk's staging
    }

    // ---- normalize + write [bs, h*128 + dv] ----
#pragma unroll
    for (int nt = 0; nt < 8; ++nt)
#pragma unroll
        for (int r = 0; r < 8; ++r) {
            int m = r + 8 * lanehi;
            size_t bs = (size_t)b * S_ + q0 + m;
            float v = accO[nt][r] / rsum[r];
            out[bs * (H_ * 128) + h * 128 + nt * 16 + lanelo] = f2bf(v);
        }
}

// ---------------------------------------------------------------------------
// Orchestration
// ---------------------------------------------------------------------------

static inline dim3 gemm_grid(int M, int N) {
    return dim3((N + 127) / 128, M / 128);
}

extern "C" void kernel_launch(void* const* d_in, const int* in_sizes, int n_in,
                              void* d_out, int out_size, void* d_ws, size_t ws_size,
                              hipStream_t stream) {
    (void)in_sizes; (void)n_in; (void)out_size; (void)ws_size;
    const float* x        = (const float*)d_in[0];
    const float* wq_down  = (const float*)d_in[1];
    const float* q_norm_w = (const float*)d_in[2];
    const float* wq_up    = (const float*)d_in[3];
    const float* wq_rope  = (const float*)d_in[4];
    const float* wkv_down = (const float*)d_in[5];
    const float* kv_norm_w= (const float*)d_in[6];
    const float* wkv_up   = (const float*)d_in[7];
    const float* wk_rope  = (const float*)d_in[8];
    const float* wo       = (const float*)d_in[9];

    uint8_t* ws = (uint8_t*)d_ws;
    size_t off = 0;
    auto alloc16 = [&](size_t elems) -> uint16_t* {
        uint16_t* p = (uint16_t*)(ws + off);
        off = (off + elems * 2 + 255) & ~(size_t)255;
        return p;
    };

    uint16_t* xb      = alloc16((size_t)BS_ * D_);
    uint16_t* wqd_t   = alloc16((size_t)QR_ * D_);
    uint16_t* wqu_t   = alloc16((size_t)(H_ * DN_) * QR_);
    uint16_t* wqr_t   = alloc16((size_t)(H_ * DR_) * QR_);
    uint16_t* wkvd_t  = alloc16((size_t)KVR_ * D_);
    uint16_t* wkvu_t  = alloc16((size_t)(H_ * (DN_ + DV_)) * KVR_);
    uint16_t* wkr_t   = alloc16((size_t)DR_ * D_);
    uint16_t* wo_t    = alloc16((size_t)D_ * (H_ * DV_));
    uint16_t* qc0     = alloc16((size_t)BS_ * QR_);
    uint16_t* qcb     = alloc16((size_t)BS_ * QR_);
    uint16_t* kvc0    = alloc16((size_t)BS_ * KVR_);
    uint16_t* kvcb    = alloc16((size_t)BS_ * KVR_);
    uint16_t* krb     = alloc16((size_t)BS_ * DR_);
    uint16_t* krr     = alloc16((size_t)BS_ * DR_);
    uint16_t* qnb     = alloc16((size_t)BS_ * H_ * DN_);
    uint16_t* qpb     = alloc16((size_t)BS_ * H_ * DR_);
    uint16_t* qpr     = alloc16((size_t)BS_ * H_ * DR_);
    uint16_t* kvub    = alloc16((size_t)BS_ * H_ * (DN_ + DV_));
    uint16_t* vtb     = alloc16((size_t)B_ * H_ * DV_ * S_);
    uint16_t* aob     = alloc16((size_t)BS_ * H_ * DV_);

    auto blocks = [](size_t n) { return (unsigned)((n + 255) / 256); };

    // 0) convert x to bf16
    cvt_f32_bf16_kernel<<<blocks((size_t)BS_ * D_), 256, 0, stream>>>(x, xb, BS_ * D_);

    // 1) convert + transpose all weights -> Bt[N][K] bf16
    cvt_transpose_kernel<<<blocks((size_t)D_ * QR_), 256, 0, stream>>>(wq_down, wqd_t, D_, QR_);
    cvt_transpose_kernel<<<blocks((size_t)QR_ * H_ * DN_), 256, 0, stream>>>(wq_up, wqu_t, QR_, H_ * DN_);
    cvt_transpose_kernel<<<blocks((size_t)QR_ * H_ * DR_), 256, 0, stream>>>(wq_rope, wqr_t, QR_, H_ * DR_);
    cvt_transpose_kernel<<<blocks((size_t)D_ * KVR_), 256, 0, stream>>>(wkv_down, wkvd_t, D_, KVR_);
    cvt_transpose_kernel<<<blocks((size_t)KVR_ * H_ * (DN_ + DV_)), 256, 0, stream>>>(wkv_up, wkvu_t, KVR_, H_ * (DN_ + DV_));
    cvt_transpose_kernel<<<blocks((size_t)D_ * DR_), 256, 0, stream>>>(wk_rope, wkr_t, D_, DR_);
    cvt_transpose_kernel<<<blocks((size_t)(H_ * DV_) * D_), 256, 0, stream>>>(wo, wo_t, H_ * DV_, D_);

    // 2) down projections + k_rope projection
    gemm_bf16_kernel<true><<<gemm_grid(BS_, QR_), 256, 0, stream>>>(xb, wqd_t, qc0, BS_, QR_, D_);
    gemm_bf16_kernel<true><<<gemm_grid(BS_, KVR_), 256, 0, stream>>>(xb, wkvd_t, kvc0, BS_, KVR_, D_);
    gemm_bf16_kernel<true><<<gemm_grid(BS_, DR_), 256, 0, stream>>>(xb, wkr_t, krb, BS_, DR_, D_);

    // 3) RMS norms
    rmsnorm_kernel<<<BS_, 256, 0, stream>>>(qc0, q_norm_w, qcb, QR_);
    rmsnorm_kernel<<<BS_, 256, 0, stream>>>(kvc0, kv_norm_w, kvcb, KVR_);

    // 4) up projections
    gemm_bf16_kernel<true><<<gemm_grid(BS_, H_ * DN_), 256, 0, stream>>>(qcb, wqu_t, qnb, BS_, H_ * DN_, QR_);
    gemm_bf16_kernel<true><<<gemm_grid(BS_, H_ * DR_), 256, 0, stream>>>(qcb, wqr_t, qpb, BS_, H_ * DR_, QR_);
    gemm_bf16_kernel<true><<<gemm_grid(BS_, H_ * (DN_ + DV_)), 256, 0, stream>>>(kvcb, wkvu_t, kvub, BS_, H_ * (DN_ + DV_), KVR_);

    // 5) RoPE (q_pe per-head, k_rope shared)
    rope_kernel<<<blocks((size_t)BS_ * H_ * 32), 256, 0, stream>>>(qpb, qpr, H_, BS_ * H_ * 32);
    rope_kernel<<<blocks((size_t)BS_ * 32), 256, 0, stream>>>(krb, krr, 1, BS_ * 32);

    // 6) V cache transpose [b,h,dv,s]
    vtrans_kernel<<<blocks((size_t)B_ * H_ * DV_ * S_), 256, 0, stream>>>(kvub, vtb);

    // 7) causal flash attention: 1 block per 128 queries of one (b,h)
    mla_attn_kernel<<<B_ * H_ * (S_ / 128), 256, 0, stream>>>(qnb, qpr, kvub, krr, vtb, aob);

    // 8) output projection -> f32 d_out
    gemm_bf16_kernel<false><<<gemm_grid(BS_, D_), 256, 0, stream>>>(aob, wo_t, d_out, BS_, D_, H_ * DV_);
}
